// LSTMmodel_455266534125
// MI455X (gfx1250) — matrix operand order repaired and checked by reading, not compiled, not run on hardware
//
#include <hip/hip_runtime.h>

#define IN_DIM 512
#define HID    1024
#define BATCH  64
#define SEQ_T  512
#define KTOT   (IN_DIM + HID)   // 1536
#define GATES  (4 * HID)        // 4096
#define NCHUNK (KTOT / 32)      // 48 K-chunks of 32
#define XCHUNK (IN_DIM / 32)    // first 16 chunks come from x, rest from h

typedef __attribute__((ext_vector_type(16))) __bf16 v16bf;
typedef __attribute__((ext_vector_type(8)))  __bf16 v8bf;
typedef __attribute__((ext_vector_type(8)))  float  v8f;

__device__ __forceinline__ __bf16 f2bf(float f) {
  unsigned u = __builtin_bit_cast(unsigned, f);
  u += 0x7FFFu + ((u >> 16) & 1u);           // round-to-nearest-even
  unsigned short s = (unsigned short)(u >> 16);
  return __builtin_bit_cast(__bf16, s);
}
__device__ __forceinline__ float bf2f(__bf16 b) {
  unsigned short s = __builtin_bit_cast(unsigned short, b);
  unsigned u = ((unsigned)s) << 16;
  return __builtin_bit_cast(float, u);
}
__device__ __forceinline__ float sigmoid_fast(float x) {
  return 1.0f / (1.0f + __expf(-x));
}
__device__ __forceinline__ float tanh_fast(float x) {
  float e = __expf(-2.0f * x);
  return (1.0f - e) / (1.0f + e);
}

// uniform base + 32-bit per-lane element offsets -> GVS-mode global_load_b128 pairs
__device__ __forceinline__ v16bf load16(const __bf16* __restrict__ base,
                                        unsigned o0, unsigned o1) {
  union { v16bf v; v8bf h[2]; } u;
  u.h[0] = *(const v8bf*)(base + o0);
  u.h[1] = *(const v8bf*)(base + o1);
  return u.v;
}

// ---------------- one-time prep kernels ----------------

__global__ void k_cvt_x(const float* __restrict__ src, __bf16* __restrict__ dst, int n) {
  for (int i = blockIdx.x * blockDim.x + threadIdx.x; i < n; i += gridDim.x * blockDim.x)
    dst[i] = f2bf(src[i]);
}

__global__ void k_pack_w(const float* __restrict__ wih, const float* __restrict__ whh,
                         __bf16* __restrict__ wc) {
  int n = GATES * KTOT;
  for (int i = blockIdx.x * blockDim.x + threadIdx.x; i < n; i += gridDim.x * blockDim.x) {
    int row = i / KTOT;
    int k   = i - row * KTOT;
    float v = (k < IN_DIM) ? wih[(size_t)row * IN_DIM + k]
                           : whh[(size_t)row * HID + (k - IN_DIM)];
    wc[i] = f2bf(v);
  }
}

__global__ void k_bias(const float* __restrict__ bih, const float* __restrict__ bhh,
                       float* __restrict__ bsum) {
  int i = blockIdx.x * blockDim.x + threadIdx.x;
  if (i < GATES) bsum[i] = bih[i] + bhh[i];
}

__global__ void k_zero_state(__bf16* __restrict__ h0, float* __restrict__ c) {
  int n = BATCH * HID;
  for (int i = blockIdx.x * blockDim.x + threadIdx.x; i < n; i += gridDim.x * blockDim.x) {
    h0[i] = __builtin_bit_cast(__bf16, (unsigned short)0);
    c[i]  = 0.0f;
  }
}

// ---------------- fused LSTM step ----------------
// Block = one hidden tile nt (16 units), 512 threads = 16 waves.
// Wave wib: mt = wib>>2 (batch tile), g = wib&3 (gate). Each wave: ONE 16x16 gate
// tile, 48-chunk K loop, 1 WMMA/chunk, software-pipelined loads.
// Accumulators exchanged through LDS, then 512 threads do the fused cell update.

__global__ __launch_bounds__(512) void lstm_step(
    const __bf16* __restrict__ xb,    // [B, T, IN_DIM] bf16
    const __bf16* __restrict__ Wc,    // [4H, KTOT] bf16
    const float*  __restrict__ bsum,  // [4H]
    const __bf16* __restrict__ h_in,  // [B, HID] bf16
    __bf16*       __restrict__ h_out, // [B, HID] bf16
    float*        __restrict__ c,     // [B, HID] f32 row-major, in place
    int t)
{
  __shared__ float gx[16 * 32 * 8];   // 16 KB: [wave][lane][8]

  const int tid  = threadIdx.x;
  const int lane = tid & 31;
  const int wib  = tid >> 5;           // 0..15
  const int mt   = wib >> 2;           // batch tile 0..3
  const int g    = wib & 3;            // gate 0..3
  const int nt   = blockIdx.x;         // hidden tile 0..63
  const int l15  = lane & 15;
  const int hiA  = (lane >> 4) << 3;   // A frag: +0 / +8
  const int hiB  = (lane >> 4) << 3;   // B frag: +0 / +8

  // 32-bit per-lane element offsets from uniform bases
  const int arow = (mt << 4) + l15;
  const unsigned xoff = ((unsigned)arow * SEQ_T + (unsigned)t) * IN_DIM + hiA;
  const unsigned hoff = (unsigned)arow * HID + hiA;
  const unsigned woff = (unsigned)(g * HID + (nt << 4) + l15) * KTOT + hiB;

  v8f acc = {};

  // pipeline prologue: chunk 0 (always x)
  v16bf aC = load16(xb, xoff, xoff + 16);
  v16bf bC = load16(Wc, woff, woff + 16);

#pragma unroll 4
  for (int kc = 1; kc < NCHUNK; ++kc) {
    const unsigned ko = kc * 32u;
    const __bf16* ab  = (kc < XCHUNK) ? xb : h_in;                      // uniform select
    const unsigned ao = (kc < XCHUNK) ? (xoff + ko) : (hoff + ko - IN_DIM);
    v16bf aN = load16(ab, ao, ao + 16);
    v16bf bN = load16(Wc, woff + ko, woff + ko + 16);

    acc = __builtin_amdgcn_wmma_f32_16x16x32_bf16(
        false, aC, false, bC, (short)0, acc, false, false);
    aC = aN;
    bC = bN;
  }
  acc = __builtin_amdgcn_wmma_f32_16x16x32_bf16(
      false, aC, false, bC, (short)0, acc, false, false);

  // ---- exchange accumulators through LDS (contiguous b128 stores) ----
  *(v8f*)&gx[((wib << 5) + lane) << 3] = acc;
  __syncthreads();

  // ---- fused cell update: 1024 elements (64 rows x 16 cols), 2 per thread ----
#pragma unroll
  for (int e = tid; e < BATCH * 16; e += 512) {
    const int row  = e >> 4;           // batch row 0..63
    const int colL = e & 15;           // column within hidden tile
    const int r    = row & 15;
    const int mt2  = row >> 4;
    const int hi   = r >> 3;
    const int v    = r & 7;
    const int ln   = colL + (hi << 4);
    const int b0   = (((mt2 << 2) << 5) + (ln)) * 8 + v;   // gate 0 slot

    const int col = (nt << 4) + colL;
    float iv = sigmoid_fast(gx[b0]             + bsum[0 * HID + col]);
    float fv = sigmoid_fast(gx[b0 + 1 * 256]   + bsum[1 * HID + col]);
    float gv = tanh_fast   (gx[b0 + 2 * 256]   + bsum[2 * HID + col]);
    float ov = sigmoid_fast(gx[b0 + 3 * 256]   + bsum[3 * HID + col]);

    const size_t idx = (size_t)row * HID + col;
    const float cv = fv * c[idx] + iv * gv;
    c[idx] = cv;
    h_out[idx] = f2bf(ov * tanh_fast(cv));
  }
}

// ---------------- head: out[b] = sigmoid(h[b,:] . V + b) ----------------
__global__ void k_head(const __bf16* __restrict__ h, const float* __restrict__ V,
                       const float* __restrict__ bias, float* __restrict__ out) {
  int b = threadIdx.x;
  if (b < BATCH) {
    float s = 0.0f;
    const __bf16* hr = h + (size_t)b * HID;
    for (int n = 0; n < HID; ++n) s += bf2f(hr[n]) * V[n];
    s += bias[0];
    out[b] = sigmoid_fast(s);
  }
}

extern "C" void kernel_launch(void* const* d_in, const int* in_sizes, int n_in,
                              void* d_out, int out_size, void* d_ws, size_t ws_size,
                              hipStream_t stream) {
  const float* x    = (const float*)d_in[0];
  const float* W_ih = (const float*)d_in[1];
  const float* W_hh = (const float*)d_in[2];
  const float* b_ih = (const float*)d_in[3];
  const float* b_hh = (const float*)d_in[4];
  const float* V    = (const float*)d_in[5];
  const float* bias = (const float*)d_in[6];

  char* ws = (char*)d_ws;
  size_t off = 0;
  auto carve = [&](size_t bytes) -> void* {
    void* p = ws + off;
    off += (bytes + 255) & ~(size_t)255;
    return p;
  };
  __bf16* xbuf = (__bf16*)carve((size_t)BATCH * SEQ_T * IN_DIM * 2); // 32 MB
  __bf16* Wc   = (__bf16*)carve((size_t)GATES * KTOT * 2);           // 12.6 MB
  float*  bsum = (float*) carve((size_t)GATES * 4);
  __bf16* hA   = (__bf16*)carve((size_t)BATCH * HID * 2);
  __bf16* hB   = (__bf16*)carve((size_t)BATCH * HID * 2);
  float*  c    = (float*) carve((size_t)BATCH * HID * 4);

  k_cvt_x     <<<2048, 256, 0, stream>>>(x, xbuf, BATCH * SEQ_T * IN_DIM);
  k_pack_w    <<<2048, 256, 0, stream>>>(W_ih, W_hh, Wc);
  k_bias      <<<(GATES + 255) / 256, 256, 0, stream>>>(b_ih, b_hh, bsum);
  k_zero_state<<<256, 256, 0, stream>>>(hA, c);

  __bf16* hin = hA;
  __bf16* hout = hB;
  for (int t = 0; t < SEQ_T; ++t) {
    lstm_step<<<64, 512, 0, stream>>>(xbuf, Wc, bsum, hin, hout, c, t);
    __bf16* tmp = hin; hin = hout; hout = tmp;
  }
  k_head<<<1, 64, 0, stream>>>(hin, V, bias, (float*)d_out);
}
